// AttentionBlock_72310069396112
// MI455X (gfx1250) — matrix operand order repaired and checked by reading, not compiled, not run on hardware
//
#include <hip/hip_runtime.h>

// AttentionBlock for MI455X (gfx1250): bf16 WMMA GEMMs + flash attention.
// b=8, c=256, n=64*64=4096, groups=8.

#define BB 8
#define CC 256
#define NN 4096
#define OO 768        // 3*C
#define GG 8
#define CPG 32        // channels per group
#define EPSV 1e-5f
#define QSCALE 0.0625f  // 1/sqrt(256)

typedef __attribute__((ext_vector_type(16))) __bf16 v16bf;
typedef __attribute__((ext_vector_type(8)))  __bf16 v8bf;
typedef __attribute__((ext_vector_type(8)))  float  v8f;
typedef __attribute__((ext_vector_type(4)))  float  v4f;

__device__ __forceinline__ __bf16 f2bf(float f) {
  union { float f; unsigned u; } x; x.f = f;
  unsigned r = x.u + 0x7FFFu + ((x.u >> 16) & 1u);   // round-nearest-even
  unsigned short h = (unsigned short)(r >> 16);
  __bf16 o; __builtin_memcpy(&o, &h, 2);
  return o;
}

__device__ __forceinline__ v8f wmma_bf(v16bf a, v16bf b, v8f c) {
  return __builtin_amdgcn_wmma_f32_16x16x32_bf16(
      /*neg_a=*/false, a, /*neg_b=*/false, b,
      /*c_mod=*/(short)0, c, /*reuse_a=*/false, /*reuse_b=*/false);
}

// A-operand (16x32, row-major K-contiguous source):
// lane holds row = lane%16; elems[0..15] = K[k0+half*16 ..] (contiguous)
__device__ __forceinline__ v16bf load_a(const __bf16* row, int k0, int half) {
  v8bf lo = *(const v8bf*)(row + k0 + half * 16);
  v8bf hi = *(const v8bf*)(row + k0 + half * 16 + 8);
  v16bf a;
#pragma unroll
  for (int i = 0; i < 8; ++i) { a[i] = lo[i]; a[i + 8] = hi[i]; }
  return a;
}

// ---------------- weight conversion ----------------
__global__ void k_wconv(const float* __restrict__ qkvw, const float* __restrict__ projw,
                        __bf16* __restrict__ wq, __bf16* __restrict__ wp) {
  int i = blockIdx.x * blockDim.x + threadIdx.x;
  if (i < OO * CC) wq[i] = f2bf(qkvw[i]);
  if (i < CC * CC) wp[i] = f2bf(projw[i]);
}

// ---------------- GroupNorm stats: one block per (b, group) ----------------
__global__ void k_gnstats(const float* __restrict__ x, float* __restrict__ stats) {
  const int b = blockIdx.x / GG, g = blockIdx.x % GG;
  const float* p = x + ((size_t)b * CC + (size_t)g * CPG) * NN;  // contiguous 131072 floats
  const int M = CPG * NN;
  float s = 0.f, sq = 0.f;
  for (int i = threadIdx.x * 4; i < M; i += blockDim.x * 4) {
    v4f v = *(const v4f*)(p + i);
    s  += v[0] + v[1] + v[2] + v[3];
    sq += v[0]*v[0] + v[1]*v[1] + v[2]*v[2] + v[3]*v[3];
  }
  __shared__ float ss[256], sqq[256];
  ss[threadIdx.x] = s; sqq[threadIdx.x] = sq;
  __syncthreads();
  for (int st = 128; st > 0; st >>= 1) {
    if ((int)threadIdx.x < st) { ss[threadIdx.x] += ss[threadIdx.x + st]; sqq[threadIdx.x] += sqq[threadIdx.x + st]; }
    __syncthreads();
  }
  if (threadIdx.x == 0) {
    float mean = ss[0] / (float)M;
    float var  = sqq[0] / (float)M - mean * mean;
    stats[blockIdx.x * 2]     = mean;
    stats[blockIdx.x * 2 + 1] = rsqrtf(var + EPSV);
  }
}

// ---------------- GN apply + transpose to [b][n][c] bf16 ----------------
__global__ void k_gnapply(const float* __restrict__ x, const float* __restrict__ gnw,
                          const float* __restrict__ gnb, const float* __restrict__ stats,
                          __bf16* __restrict__ xn) {
  size_t i = (size_t)blockIdx.x * blockDim.x + threadIdx.x;  // over B*C*N
  int b   = (int)(i / ((size_t)CC * NN));
  int rem = (int)(i % ((size_t)CC * NN));
  int ch = rem / NN, pix = rem % NN;
  int g = ch / CPG;
  float mean = stats[(b * GG + g) * 2];
  float rstd = stats[(b * GG + g) * 2 + 1];
  float v = (x[i] - mean) * rstd * gnw[ch] + gnb[ch];
  xn[((size_t)b * NN + pix) * CC + ch] = f2bf(v);
}

// ---------------- QKV GEMM: D[o,n] = Wqkv[o,:] . xn[n,:] ----------------
__global__ void __launch_bounds__(128) k_qkv(const __bf16* __restrict__ wq,
                                             const float* __restrict__ qkvb,
                                             const __bf16* __restrict__ xn,
                                             __bf16* __restrict__ qb,
                                             __bf16* __restrict__ kbuf,
                                             __bf16* __restrict__ vb) {
  int wid  = blockIdx.x * 4 + (threadIdx.x >> 5);
  int lane = threadIdx.x & 31;
  int half = lane >> 4, l16 = lane & 15;
  int b  = wid / (48 * 256);
  int r  = wid % (48 * 256);
  int ot = r / 256, nt = r % 256;

  const __bf16* arow  = wq + (size_t)(ot * 16 + l16) * CC;
  const __bf16* bbase = xn + ((size_t)b * NN + nt * 16 + l16) * CC + half * 16;
  v8f acc = {};
#pragma unroll
  for (int kk = 0; kk < 8; ++kk) {
    v16bf a  = load_a(arow, kk * 32, half);
    v16bf bm = *(const v16bf*)(bbase + kk * 32);
    acc = wmma_bf(a, bm, acc);
  }
  int nn = nt * 16 + l16;
#pragma unroll
  for (int v = 0; v < 8; ++v) {
    int o = ot * 16 + v + 8 * half;
    float val = acc[v] + qkvb[o];
    if (o < CC)            qb  [((size_t)b * NN + nn) * CC + o]            = f2bf(val * QSCALE);
    else if (o < 2 * CC)   kbuf[((size_t)b * NN + nn) * CC + (o - CC)]     = f2bf(val);
    else                   vb  [((size_t)b * CC + (o - 2 * CC)) * NN + nn] = f2bf(val);
  }
}

// ---------------- Flash attention ----------------
// Block = 4 waves x 32. Each wave: its own 16-query tile, full 256 output channels.
// Online softmax; P re-laid-out C/D -> A via 1KB/wave LDS, ordered by s_wait_dscnt.
__global__ void __launch_bounds__(128) k_attn(const __bf16* __restrict__ qb,
                                              const __bf16* __restrict__ kbuf,
                                              const __bf16* __restrict__ vb,
                                              __bf16* __restrict__ ob) {
  __shared__ __align__(32) __bf16 psh[4][16 * 32];
  int wv   = threadIdx.x >> 5;
  int lane = threadIdx.x & 31;
  int half = lane >> 4, l16 = lane & 15;
  int b  = blockIdx.x >> 6;               // /64
  int qg = blockIdx.x & 63;               // 64-query group
  int q0 = qg * 64 + wv * 16;             // this wave's query tile base

  const __bf16* qrow = qb   + ((size_t)b * NN + q0 + l16) * CC;
  const __bf16* kbb  = kbuf + (size_t)b * NN * CC;
  const __bf16* vbb  = vb   + (size_t)b * CC * NN;

  v8f acc[16] = {};
  float mrow[8], lrow[8];
#pragma unroll
  for (int v = 0; v < 8; ++v) { mrow[v] = -1e30f; lrow[v] = 0.f; }

  for (int m0 = 0; m0 < NN; m0 += 32) {
    // S = Q . K^T for 32 keys (two 16-col tiles), K-dim = c = 256
    v8f s0 = {}, s1 = {};
#pragma unroll
    for (int kk = 0; kk < 8; ++kk) {
      v16bf a   = load_a(qrow, kk * 32, half);
      v16bf kb0 = *(const v16bf*)(kbb + (size_t)(m0 + l16) * CC + kk * 32 + half * 16);
      v16bf kb1 = *(const v16bf*)(kbb + (size_t)(m0 + 16 + l16) * CC + kk * 32 + half * 16);
      s0 = wmma_bf(a, kb0, s0);
      s1 = wmma_bf(a, kb1, s1);
    }
    // online softmax per row-register (row = v + 8*half, spread over 16 lanes)
#pragma unroll
    for (int v = 0; v < 8; ++v) {
      float a0 = s0[v], a1 = s1[v];
      float rmax = fmaxf(a0, a1);
#pragma unroll
      for (int off = 1; off < 16; off <<= 1) rmax = fmaxf(rmax, __shfl_xor(rmax, off, 32));
      float mnew = fmaxf(mrow[v], rmax);
      float corr = __expf(mrow[v] - mnew);
      float p0 = __expf(a0 - mnew), p1 = __expf(a1 - mnew);
      float rs = p0 + p1;
#pragma unroll
      for (int off = 1; off < 16; off <<= 1) rs += __shfl_xor(rs, off, 32);
      lrow[v] = lrow[v] * corr + rs;
      mrow[v] = mnew;
#pragma unroll
      for (int t = 0; t < 16; ++t) acc[t][v] *= corr;
      int rr = v + 8 * half;
      psh[wv][rr * 32 + l16]      = f2bf(p0);
      psh[wv][rr * 32 + 16 + l16] = f2bf(p1);
    }
    asm volatile("s_wait_dscnt 0" ::: "memory");   // LDS RAW: P stores -> A-frag loads
    v16bf pf = load_a(&psh[wv][l16 * 32], 0, half);
    // O += P . V^T  (K-dim = 32 keys), 16 channel tiles
#pragma unroll
    for (int t = 0; t < 16; ++t) {
      int ch = t * 16 + l16;
      v16bf vf = *(const v16bf*)(vbb + (size_t)ch * NN + m0 + half * 16);
      acc[t] = wmma_bf(pf, vf, acc[t]);
    }
  }
  // normalize by row-sum and store O as [b][n][c] bf16
#pragma unroll
  for (int v = 0; v < 8; ++v) {
    float inv = 1.f / lrow[v];
    int qrow_i = q0 + v + 8 * half;
    __bf16* orow = ob + ((size_t)b * NN + qrow_i) * CC + l16;
#pragma unroll
    for (int t = 0; t < 16; ++t) orow[t * 16] = f2bf(acc[t][v] * inv);
  }
}

// ---------------- proj GEMM + bias + residual, f32 out [b][c][n] ----------------
__global__ void __launch_bounds__(128) k_proj(const __bf16* __restrict__ wp,
                                              const float* __restrict__ pb,
                                              const __bf16* __restrict__ ob,
                                              const float* __restrict__ x,
                                              float* __restrict__ out) {
  int wid  = blockIdx.x * 4 + (threadIdx.x >> 5);
  int lane = threadIdx.x & 31;
  int half = lane >> 4, l16 = lane & 15;
  int b  = wid / (16 * 256);
  int r  = wid % (16 * 256);
  int ot = r / 256, nt = r % 256;

  const __bf16* arow  = wp + (size_t)(ot * 16 + l16) * CC;
  const __bf16* bbase = ob + ((size_t)b * NN + nt * 16 + l16) * CC + half * 16;
  v8f acc = {};
#pragma unroll
  for (int kk = 0; kk < 8; ++kk) {
    v16bf a  = load_a(arow, kk * 32, half);
    v16bf bm = *(const v16bf*)(bbase + kk * 32);
    acc = wmma_bf(a, bm, acc);
  }
  int nn = nt * 16 + l16;
#pragma unroll
  for (int v = 0; v < 8; ++v) {
    int och = ot * 16 + v + 8 * half;
    size_t idx = ((size_t)b * CC + och) * NN + nn;
    out[idx] = x[idx] + acc[v] + pb[och];
  }
}

extern "C" void kernel_launch(void* const* d_in, const int* in_sizes, int n_in,
                              void* d_out, int out_size, void* d_ws, size_t ws_size,
                              hipStream_t stream) {
  const float* x     = (const float*)d_in[0];
  const float* gnw   = (const float*)d_in[1];
  const float* gnb   = (const float*)d_in[2];
  const float* qkvw  = (const float*)d_in[3];
  const float* qkvb  = (const float*)d_in[4];
  const float* projw = (const float*)d_in[5];
  const float* projb = (const float*)d_in[6];
  float* out = (float*)d_out;

  char* ws = (char*)d_ws;
  const size_t SZ = (size_t)BB * NN * CC * 2;   // one bf16 [b][*][*] tensor
  float*  stats = (float*)ws;                                   // 512 B
  __bf16* xn = (__bf16*)(ws + 4096);
  __bf16* qb = (__bf16*)(ws + 4096 + 1 * SZ);
  __bf16* kb = (__bf16*)(ws + 4096 + 2 * SZ);
  __bf16* vb = (__bf16*)(ws + 4096 + 3 * SZ);
  __bf16* ob = (__bf16*)(ws + 4096 + 4 * SZ);
  __bf16* wq = (__bf16*)(ws + 4096 + 5 * SZ);
  __bf16* wp = (__bf16*)(ws + 4096 + 5 * SZ + (size_t)OO * CC * 2);

  k_wconv  <<<(OO * CC + 255) / 256, 256, 0, stream>>>(qkvw, projw, wq, wp);
  k_gnstats<<<BB * GG,              256, 0, stream>>>(x, stats);
  k_gnapply<<<(BB * CC * NN) / 256, 256, 0, stream>>>(x, gnw, gnb, stats, xn);
  k_qkv    <<<BB * 48 * 256 / 4,    128, 0, stream>>>(wq, qkvb, xn, qb, kb, vb);
  k_attn   <<<BB * 64,              128, 0, stream>>>(qb, kb, vb, ob);
  k_proj   <<<BB * 16 * 256 / 4,    128, 0, stream>>>(wp, projb, ob, x, out);
}
